// GatedDeltaRuleAttention_40278203302388
// MI455X (gfx1250) — compile-verified, matrix operand order checked by reading, not run on hardware
//
#include <hip/hip_runtime.h>
#include <hip/hip_bf16.h>

// ---------------- problem constants ----------------
#define B_SZ   2
#define S_SZ   4096
#define D_SZ   2048
#define HK_N   16
#define HV_N   32
#define DK_N   128
#define DV_N   128
#define NTOK   (B_SZ * S_SZ)          // 8192
#define KEYDIM (HK_N * DK_N)          // 2048
#define VALDIM (HV_N * DV_N)          // 4096
#define QKVZ_N (2 * KEYDIM + 2 * VALDIM) // 12288
#define CONV_N (2 * KEYDIM + VALDIM)  // 8192
#define CHUNK  64
#define NCHUNK (S_SZ / CHUNK)         // 64

typedef __bf16 bf16_t;
typedef __attribute__((ext_vector_type(16))) __bf16 v16bf;
typedef __attribute__((ext_vector_type(8)))  float  v8f;

// ---- TDM support (CDNA5 tensor data mover) ----
#if defined(__has_builtin)
#if __has_builtin(__builtin_amdgcn_tensor_load_to_lds) && \
    __has_builtin(__builtin_amdgcn_tensor_store_from_lds) && \
    __has_builtin(__builtin_amdgcn_s_wait_tensorcnt)
#define HAVE_TDM 1
#endif
#endif
#ifndef HAVE_TDM
#define HAVE_TDM 0
#endif

typedef unsigned int u32x4 __attribute__((ext_vector_type(4)));
typedef int i32x4 __attribute__((ext_vector_type(4)));
typedef int i32x8 __attribute__((ext_vector_type(8)));

#if HAVE_TDM
// D# builder per CDNA5 ISA.
__device__ __forceinline__ void tdm_desc(u32x4& g0, i32x8& g1,
                                         const void* gaddr, unsigned lds_off, int dsize_code,
                                         int tile_d0, int tile_d1,
                                         unsigned tensor_d0, unsigned tensor_d1,
                                         unsigned long long stride_elems,
                                         int pad_interval_code, int pad_amount_code) {
  const unsigned long long ga = (unsigned long long)gaddr;
  g0[0] = 1u;
  g0[1] = lds_off;
  g0[2] = (unsigned)(ga & 0xffffffffu);
  g0[3] = (unsigned)((ga >> 32) & 0x01ffffffu) | (2u << 30);
  unsigned w0 = ((unsigned)dsize_code << 16);
  if (pad_amount_code >= 0)
    w0 |= (1u << 20) | ((unsigned)pad_interval_code << 22) | ((unsigned)pad_amount_code << 25);
  g1[0] = (int)w0;
  g1[1] = (int)((tensor_d0 & 0xffffu) << 16);
  g1[2] = (int)((tensor_d0 >> 16) | ((tensor_d1 & 0xffffu) << 16));
  g1[3] = (int)((tensor_d1 >> 16) | ((unsigned)tile_d0 << 16));
  g1[4] = (int)((unsigned)tile_d1 & 0xffffu);
  g1[5] = (int)(stride_elems & 0xffffffffu);
  g1[6] = (int)((stride_elems >> 32) & 0xffffu);
  g1[7] = 0;
}

__device__ __forceinline__ void tdm_load_2d(const void* gaddr, unsigned lds_off, int dsize_code,
                                            int tile_d0, int tile_d1,
                                            unsigned tensor_d0, unsigned tensor_d1,
                                            unsigned long long stride_elems,
                                            int pic, int pac) {
  u32x4 g0; i32x8 g1;
  tdm_desc(g0, g1, gaddr, lds_off, dsize_code, tile_d0, tile_d1, tensor_d0, tensor_d1,
           stride_elems, pic, pac);
  const i32x4 z4 = (i32x4){0, 0, 0, 0};
#if __clang_major__ >= 23
  const i32x8 z8 = (i32x8){0, 0, 0, 0, 0, 0, 0, 0};
  __builtin_amdgcn_tensor_load_to_lds(g0, g1, z4, z4, z8, 0);
#else
  __builtin_amdgcn_tensor_load_to_lds(g0, g1, z4, z4, 0);
#endif
}

__device__ __forceinline__ void tdm_store_2d(const void* gaddr, unsigned lds_off, int dsize_code,
                                             int tile_d0, int tile_d1,
                                             unsigned tensor_d0, unsigned tensor_d1,
                                             unsigned long long stride_elems) {
  u32x4 g0; i32x8 g1;
  tdm_desc(g0, g1, gaddr, lds_off, dsize_code, tile_d0, tile_d1, tensor_d0, tensor_d1,
           stride_elems, 0, -1);
  const i32x4 z4 = (i32x4){0, 0, 0, 0};
#if __clang_major__ >= 23
  const i32x8 z8 = (i32x8){0, 0, 0, 0, 0, 0, 0, 0};
  __builtin_amdgcn_tensor_store_from_lds(g0, g1, z4, z4, z8, 0);
#else
  __builtin_amdgcn_tensor_store_from_lds(g0, g1, z4, z4, 0);
#endif
}
#endif

union V8B { int4 v; bf16_t h[8]; };

__device__ __forceinline__ float clip30(float x) { return fminf(fmaxf(x, -30.f), 30.f); }
__device__ __forceinline__ float sigmoidf_(float x) { return 1.f / (1.f + expf(-x)); }

__device__ __forceinline__ v8f wmma_bf16(v16bf a, v16bf b, v8f c) {
  return __builtin_amdgcn_wmma_f32_16x16x32_bf16(false, a, false, b, (short)0, c, false, false);
}

// ---- fragment loader (wave32 16-bit A/B layout per CDNA5 ISA 7.12.2) ----
__device__ __forceinline__ v16bf ld_frag(const bf16_t* buf, int ld, int r0, int k0) {
  const int lane = threadIdx.x & 31;
  const bf16_t* p = buf + (r0 + (lane & 15)) * ld + k0 + ((lane & 16) ? 8 : 0);
  v16bf f;
#pragma unroll
  for (int v = 0; v < 8; ++v) {
    const int k = (v < 4) ? (2 * v) : (16 + 2 * (v - 4));
    f[2 * v]     = p[k];
    f[2 * v + 1] = p[k + 1];
  }
  return f;
}

// ---------------- kernels ----------------

__global__ void __launch_bounds__(256)
f32_to_bf16_kernel(const float* __restrict__ in, bf16_t* __restrict__ out, long n) {
  long i = (long)blockIdx.x * blockDim.x + threadIdx.x;
  const long stride = (long)gridDim.x * blockDim.x;
  for (; i < n; i += stride) out[i] = (bf16_t)in[i];
}

// C[M,N] = A[M,K](bf16) * B[K,N](bf16); f32 or bf16 out.
template <bool ALIGNED>
__global__ void __launch_bounds__(256)
gemm_bf16_kernel(const bf16_t* __restrict__ A, const bf16_t* __restrict__ Bm,
                 float* __restrict__ Cf, bf16_t* __restrict__ Cb,
                 int M, int N, int K) {
  __shared__ bf16_t As[128][40];
  __shared__ bf16_t BsT[128][40];
  const int tid  = threadIdx.x;
  const int wid  = tid >> 5;
  const int lane = tid & 31;
  const int m0b = blockIdx.y * 128;
  const int n0b = blockIdx.x * 128;
  const int wm = (wid & 3) * 32;
  const int wn = (wid >> 2) * 64;

  v8f acc[2][4];
#pragma unroll
  for (int i = 0; i < 2; ++i)
#pragma unroll
    for (int j = 0; j < 4; ++j) acc[i][j] = (v8f){0.f,0.f,0.f,0.f,0.f,0.f,0.f,0.f};

  const int kTiles = (K + 31) >> 5;
  for (int kt = 0; kt < kTiles; ++kt) {
    const int k0 = kt << 5;
    if (ALIGNED) {
#if HAVE_TDM
      if (wid == 0) {
        tdm_load_2d(A + (size_t)m0b * K + k0, (unsigned)(size_t)&As[0][0], 1,
                    32, 128, (unsigned)(K - k0), (unsigned)(M - m0b),
                    (unsigned long long)K, 3, 3);
      }
#else
#pragma unroll
      for (int half = 0; half < 2; ++half) {
        const int e = tid * 16 + half * 8;
        const int r = e >> 5, kk = e & 31;
        V8B u;
        u.v = *(const int4*)(A + (size_t)(m0b + r) * K + k0 + kk);
        *(int4*)&As[r][kk] = u.v;
      }
#endif
#pragma unroll
      for (int half = 0; half < 2; ++half) {
        const int e = tid * 16 + half * 8;
        const int r = e >> 7, n = e & 127;
        V8B u;
        u.v = *(const int4*)(Bm + (size_t)(k0 + r) * N + n0b + n);
#pragma unroll
        for (int j = 0; j < 8; ++j) BsT[n + j][r] = u.h[j];
      }
      if (kt + 1 < kTiles)
        __builtin_prefetch(Bm + (size_t)(k0 + 32 + (tid & 31)) * N + n0b, 0, 1);
#if HAVE_TDM
      if (wid == 0) __builtin_amdgcn_s_wait_tensorcnt(0);
#endif
    } else {
      for (int i = tid; i < 128 * 32; i += 256) {
        const int r = i >> 5, kk = i & 31;
        const int gm = m0b + r, gk = k0 + kk;
        As[r][kk] = (gm < M && gk < K) ? A[(size_t)gm * K + gk] : (bf16_t)0.f;
      }
      for (int i = tid; i < 32 * 128; i += 256) {
        const int r = i >> 7, n = i & 127;
        const int gk = k0 + r, gn = n0b + n;
        BsT[n][r] = (gk < K && gn < N) ? Bm[(size_t)gk * N + gn] : (bf16_t)0.f;
      }
    }
    __syncthreads();
    v16bf af[2], bfr[4];
#pragma unroll
    for (int i = 0; i < 2; ++i) af[i] = ld_frag(&As[0][0], 40, wm + 16 * i, 0);
#pragma unroll
    for (int j = 0; j < 4; ++j) bfr[j] = ld_frag(&BsT[0][0], 40, wn + 16 * j, 0);
#pragma unroll
    for (int i = 0; i < 2; ++i)
#pragma unroll
      for (int j = 0; j < 4; ++j) acc[i][j] = wmma_bf16(af[i], bfr[j], acc[i][j]);
    __syncthreads();
  }

  const int lm = (lane & 16) ? 8 : 0;
  const int cn = lane & 15;
#pragma unroll
  for (int i = 0; i < 2; ++i)
#pragma unroll
    for (int j = 0; j < 4; ++j) {
      const int col = n0b + wn + 16 * j + cn;
      if (col >= N) continue;
#pragma unroll
      for (int r = 0; r < 8; ++r) {
        const int row = m0b + wm + 16 * i + lm + r;
        if (row < M) {
          const size_t idx = (size_t)row * N + col;
          if (Cb) Cb[idx] = (bf16_t)acc[i][j][r];
          else    Cf[idx] = acc[i][j][r];
        }
      }
    }
}

// causal depthwise conv (ks=4) + SiLU; 8 channels per thread, vectorized.
__global__ void __launch_bounds__(256)
conv_silu_kernel(const bf16_t* __restrict__ qkvz, const float* __restrict__ w,
                 bf16_t* __restrict__ qkvc) {
  const long idx = (long)blockIdx.x * 256 + threadIdx.x;
  if (idx >= (long)NTOK * (CONV_N / 8)) return;
  const int c8 = (int)(idx & (CONV_N / 8 - 1)) * 8;
  const long p = idx >> 10;
  const int s = (int)(p & (S_SZ - 1));
  int col0;
  if (c8 < KEYDIM) {
    col0 = (c8 >> 7) * 768 + (c8 & 127);
  } else if (c8 < 2 * KEYDIM) {
    const int c2 = c8 - KEYDIM;
    col0 = (c2 >> 7) * 768 + 128 + (c2 & 127);
  } else {
    const int c3 = c8 - 2 * KEYDIM;
    const int hv = c3 >> 7;
    col0 = (hv >> 1) * 768 + 256 + (hv & 1) * 128 + (c3 & 127);
  }
  float4 wv[8];
#pragma unroll
  for (int j = 0; j < 8; ++j) wv[j] = *(const float4*)(w + (c8 + j) * 4);
  float acc8[8];
#pragma unroll
  for (int j = 0; j < 8; ++j) acc8[j] = 0.f;
#pragma unroll
  for (int t = 0; t < 4; ++t) {
    const int sd = s - 3 + t;
    if (sd < 0) continue;
    V8B u;
    u.v = *(const int4*)(qkvz + (p - 3 + t) * QKVZ_N + col0);
    const float* wt = (const float*)&wv[0];
#pragma unroll
    for (int j = 0; j < 8; ++j) acc8[j] += wt[j * 4 + t] * (float)u.h[j];
  }
  V8B o;
#pragma unroll
  for (int j = 0; j < 8; ++j) o.h[j] = (bf16_t)(acc8[j] * sigmoidf_(acc8[j]));
  *(int4*)(qkvc + p * CONV_N + c8) = o.v;
}

__global__ void __launch_bounds__(256)
betag_kernel(const float* __restrict__ ba, const float* __restrict__ A_log,
             const float* __restrict__ dtb, float* __restrict__ beta,
             float* __restrict__ g) {
  const long idx = (long)blockIdx.x * 256 + threadIdx.x;
  if (idx >= (long)NTOK * HV_N) return;
  const int hv = (int)(idx & (HV_N - 1));
  const long pos = idx >> 5;
  const int hk = hv >> 1, gi = hv & 1;
  const float br = ba[pos * 64 + hk * 4 + gi];
  const float av = ba[pos * 64 + hk * 4 + 2 + gi];
  beta[idx] = sigmoidf_(br);
  const float x = av + dtb[hv];
  const float sp = (x > 20.f) ? x : log1pf(expf(x));
  g[idx] = -expf(A_log[hv]) * sp;
}

// ---------------- gated delta rule core ----------------
__global__ void __launch_bounds__(256)
delta_core_kernel(const bf16_t* __restrict__ qkv, const float* __restrict__ betap,
                  const float* __restrict__ gp, float* __restrict__ core) {
  extern __shared__ char smem[];
  float*  state = (float*)smem;            // 128x128 f32
  float*  Tf    = state + 128 * 128;       // 64x64 (A, then diag-inverted blocks)
  float*  uf    = Tf + 64 * 64;            // 64x128 (v stage / value_corrected / out stage)
  float*  gc    = uf + 64 * 128;           // 64
  float*  ge    = gc + 64;                 // 64  e^{g}
  float*  gei   = ge + 64;                 // 64  1/e^{g}
  float*  kd    = gei + 64;                // 64
  float*  betac = kd + 64;                 // 64
  float*  red   = betac + 64;              // 256
  float*  scal  = red + 256;               // 8
  bf16_t* sbT   = (bf16_t*)(scal + 8);     // state^T bf16 [dv][d]
  bf16_t* qb    = sbT + 128 * 128;         // [64][128]
  bf16_t* kb    = qb + 64 * 128;           // [64][128]
  bf16_t* kbb   = kb + 64 * 128;           // [64][128]  k*beta
  bf16_t* kkdT  = kbb + 64 * 128;          // [128][64]  (k*kd)^T
  bf16_t* vbT   = kkdT + 128 * 64;         // [128][64]  (v*beta)^T
  bf16_t* kgT   = vbT + 128 * 64;          // [128][64]  (k*beta*e^g)^T
  bf16_t* vpT   = kgT + 128 * 64;          // [128][64]  v'^T
  bf16_t* wb    = vpT + 128 * 64;          // [64][128]  k_cumdecay
  bf16_t* Tb    = wb + 64 * 128;           // [64][64]   T = (I-A)^-1 (row major)
  bf16_t* TbT   = Tb + 64 * 64;            // [64][64]   T transposed
  bf16_t* Ds    = TbT + 64 * 64;           // 4 x [16][32] diag blocks, K-padded
  bf16_t* SsT   = Ds + 4 * 16 * 32;        // 4 x [16][32] S^T scratch, K-padded
  bf16_t* intb  = SsT + 4 * 16 * 32;       // [64][64]   A(bf16) during R, intra after

  const int tid  = threadIdx.x;
  const int wid  = tid >> 5;
  const int lane = tid & 31;
  const int hv = blockIdx.x & 31;
  const int bb = blockIdx.x >> 5;
  const int hk = hv >> 1;
  const int row4 = tid >> 2;
  const int seg  = tid & 3;

  for (int i = tid; i < 128 * 128; i += 256) { state[i] = 0.f; sbT[i] = (bf16_t)0.f; }
  // zero Tb/TbT/Ds/SsT once: unwritten regions must stay zero across chunks
  for (int i = tid; i < 64 * 64 * 2 + 2 * 4 * 16 * 32; i += 256) Tb[i] = (bf16_t)0.f;
  __syncthreads();

  for (int c = 0; c < NCHUNK; ++c) {
    const long p0 = (long)bb * S_SZ + (long)c * CHUNK;

    if (tid == 0) {
      float run = 0.f;
      for (int i = 0; i < CHUNK; ++i) {
        run += gp[(p0 + i) * HV_N + hv];
        gc[i] = run;
        const float e = expf(clip30(run));
        ge[i] = e;
        gei[i] = 1.f / e;
        betac[i] = betap[(p0 + i) * HV_N + hv];
      }
      const float gl = gc[CHUNK - 1];
      scal[0] = expf(clip30(gl));
      for (int i = 0; i < CHUNK; ++i) kd[i] = expf(clip30(gl - gc[i]));
    }
#if HAVE_TDM
    if (wid == 0) {
      tdm_load_2d(qkv + p0 * CONV_N + hk * 128, (unsigned)(size_t)qb, 1,
                  128, 64, 128u, 64u, (unsigned long long)CONV_N, 0, -1);
      tdm_load_2d(qkv + p0 * CONV_N + KEYDIM + hk * 128, (unsigned)(size_t)kb, 1,
                  128, 64, 128u, 64u, (unsigned long long)CONV_N, 0, -1);
      tdm_load_2d(qkv + p0 * CONV_N + 2 * KEYDIM + hv * 128, (unsigned)(size_t)uf, 1,
                  128, 64, 128u, 64u, (unsigned long long)CONV_N, 0, -1);
      __builtin_amdgcn_s_wait_tensorcnt(0);
    }
#endif
    __syncthreads();

    // ---- l2-normalize q ----
    {
#if HAVE_TDM
      const bf16_t* src = qb + row4 * 128 + seg * 32;
#else
      const bf16_t* src = qkv + (p0 + row4) * CONV_N + hk * 128 + seg * 32;
#endif
      float rv[32], ss = 0.f;
#pragma unroll
      for (int u4 = 0; u4 < 4; ++u4) {
        V8B u; u.v = *(const int4*)(src + u4 * 8);
#pragma unroll
        for (int j = 0; j < 8; ++j) { float f = (float)u.h[j]; rv[u4 * 8 + j] = f; ss += f * f; }
      }
      red[tid] = ss;
      __syncthreads();
      const float s = red[row4 * 4] + red[row4 * 4 + 1] + red[row4 * 4 + 2] + red[row4 * 4 + 3];
      const float rn = rsqrtf(s + 1e-6f) * 0.08838834764831845f;
#pragma unroll
      for (int u4 = 0; u4 < 4; ++u4) {
        V8B o;
#pragma unroll
        for (int j = 0; j < 8; ++j) o.h[j] = (bf16_t)(rv[u4 * 8 + j] * rn);
        *(int4*)&qb[row4 * 128 + seg * 32 + u4 * 8] = o.v;
      }
      __syncthreads();
    }
    // ---- l2-normalize k; derive k*beta, (k*kd)^T, (k*beta*e^g)^T ----
    {
#if HAVE_TDM
      const bf16_t* src = kb + row4 * 128 + seg * 32;
#else
      const bf16_t* src = qkv + (p0 + row4) * CONV_N + KEYDIM + hk * 128 + seg * 32;
#endif
      float rv[32], ss = 0.f;
#pragma unroll
      for (int u4 = 0; u4 < 4; ++u4) {
        V8B u; u.v = *(const int4*)(src + u4 * 8);
#pragma unroll
        for (int j = 0; j < 8; ++j) { float f = (float)u.h[j]; rv[u4 * 8 + j] = f; ss += f * f; }
      }
      red[tid] = ss;
      __syncthreads();
      const float s = red[row4 * 4] + red[row4 * 4 + 1] + red[row4 * 4 + 2] + red[row4 * 4 + 3];
      const float rn = rsqrtf(s + 1e-6f);
      const float bta = betac[row4];
      const float bg = bta * ge[row4];
      const float kdr = kd[row4];
#pragma unroll
      for (int u4 = 0; u4 < 4; ++u4) {
        V8B ok, okb;
#pragma unroll
        for (int j = 0; j < 8; ++j) {
          const int d = seg * 32 + u4 * 8 + j;
          const float f = rv[u4 * 8 + j] * rn;
          ok.h[j]  = (bf16_t)f;
          okb.h[j] = (bf16_t)(f * bta);
          kkdT[d * 64 + row4] = (bf16_t)(f * kdr);
          kgT[d * 64 + row4]  = (bf16_t)(f * bg);
        }
        *(int4*)&kb[row4 * 128 + seg * 32 + u4 * 8]  = ok.v;
        *(int4*)&kbb[row4 * 128 + seg * 32 + u4 * 8] = okb.v;
      }
    }
    // ---- v -> (v*beta)^T ----
    {
#if HAVE_TDM
      const bf16_t* src = (const bf16_t*)uf + row4 * 128 + seg * 32;
#else
      const bf16_t* src = qkv + (p0 + row4) * CONV_N + 2 * KEYDIM + hv * 128 + seg * 32;
#endif
      const float bta = betac[row4];
#pragma unroll
      for (int u4 = 0; u4 < 4; ++u4) {
        V8B u; u.v = *(const int4*)(src + u4 * 8);
#pragma unroll
        for (int j = 0; j < 8; ++j)
          vbT[(seg * 32 + u4 * 8 + j) * 64 + row4] = (bf16_t)((float)u.h[j] * bta);
      }
    }
    __syncthreads();

    // ---- phase A: Tf = strict-lower( -(k*beta @ k^T) * e^{g_r}/e^{g_c} ) ----
    for (int t = wid; t < 16; t += 8) {
      const int tm = (t >> 2) * 16, tn = (t & 3) * 16;
      v8f acc = (v8f){0.f,0.f,0.f,0.f,0.f,0.f,0.f,0.f};
#pragma unroll
      for (int kk = 0; kk < 4; ++kk) {
        v16bf a = ld_frag(kbb, 128, tm, kk * 32);
        v16bf b = ld_frag(kb, 128, tn, kk * 32);
        acc = wmma_bf16(a, b, acc);
      }
      const int col = tn + (lane & 15);
      const int rb = tm + ((lane & 16) ? 8 : 0);
      const float dinv = gei[col];
      float grow[8];
#pragma unroll
      for (int r = 0; r < 8; ++r) grow[r] = ge[rb + r];
#pragma unroll
      for (int r = 0; r < 8; ++r) {
        const int row = rb + r;
        const float v = -acc[r] * grow[r] * dinv;
        Tf[row * 64 + col] = (col < row) ? v : 0.f;
      }
    }
    __syncthreads();

    // ---- phase R: blocked T = (I - A)^{-1} using WMMA ----
    // copy A -> bf16 (intb); invert the 4 diagonal 16x16 blocks warp-locally.
    {
      const int i0 = tid * 16;
      V8B a1, a2;
#pragma unroll
      for (int j = 0; j < 8; ++j) { a1.h[j] = (bf16_t)Tf[i0 + j]; a2.h[j] = (bf16_t)Tf[i0 + 8 + j]; }
      *(int4*)&intb[i0] = a1.v;
      *(int4*)&intb[i0 + 8] = a2.v;
    }
    __syncthreads();
    if (wid < 4 && lane < 16) {
      float* Bp = Tf + (wid * 16) * 64 + wid * 16;   // diag block, row stride 64
      for (int i = 1; i < 16; ++i) {
        float corr = 0.f;
        for (int m = 0; m < i; ++m) corr += Bp[i * 64 + m] * Bp[m * 64 + lane];
        if (lane < i) Bp[i * 64 + lane] += corr;     // per-wave LDS ordering
      }
    }
    __syncthreads();
    if (tid < 64) {   // emit diag blocks (+I) to Tb, TbT and padded Ds
      const int bidx = tid >> 4, r = tid & 15;
      const int rowg = bidx * 16 + r;
#pragma unroll
      for (int j = 0; j < 16; ++j) {
        const float v = (j < r) ? Tf[rowg * 64 + bidx * 16 + j] : ((j == r) ? 1.f : 0.f);
        Tb [rowg * 64 + bidx * 16 + j] = (bf16_t)v;
        TbT[(bidx * 16 + j) * 64 + rowg] = (bf16_t)v;
        Ds [bidx * 512 + r * 32 + j] = (bf16_t)v;
      }
    }
    __syncthreads();
    // off-diagonal: T[bi][bj] = Tdiag[bi] * (sum_m A[bi][m] * T[m][bj])
    for (int bi = 1; bi < 4; ++bi) {
      if (wid < bi) {
        const int bj = wid;
        v8f acc = (v8f){0.f,0.f,0.f,0.f,0.f,0.f,0.f,0.f};
#pragma unroll
        for (int kk = 0; kk < 2; ++kk) {
          v16bf a = ld_frag(intb, 64, bi * 16, kk * 32);
          v16bf b = ld_frag(TbT, 64, bj * 16, kk * 32);
          acc = wmma_bf16(a, b, acc);
        }
        const int col = lane & 15;
        const int rbase = (lane & 16) ? 8 : 0;
#pragma unroll
        for (int r = 0; r < 8; ++r) SsT[wid * 512 + col * 32 + rbase + r] = (bf16_t)acc[r];
        v16bf a2 = ld_frag(Ds + bi * 512, 32, 0, 0);
        v16bf b2 = ld_frag(SsT + wid * 512, 32, 0, 0);
        v8f tres = wmma_bf16(a2, b2, (v8f){0.f,0.f,0.f,0.f,0.f,0.f,0.f,0.f});
#pragma unroll
        for (int r = 0; r < 8; ++r) {
          const int row = bi * 16 + rbase + r;
          const int cg = bj * 16 + col;
          Tb [row * 64 + cg] = (bf16_t)tres[r];
          TbT[cg * 64 + row] = (bf16_t)tres[r];
        }
      }
      __syncthreads();
    }

    // ---- phase U: uf = T @ (v*beta) ; wb = T @ (k*beta*e^g) ----
    for (int t = wid; t < 64; t += 8) {
      const int isU = (t < 32);
      const int tt = isU ? t : t - 32;
      const int tm = (tt >> 3) * 16, tn = (tt & 7) * 16;
      const bf16_t* Bp = isU ? vbT : kgT;
      v8f acc = (v8f){0.f,0.f,0.f,0.f,0.f,0.f,0.f,0.f};
#pragma unroll
      for (int kk = 0; kk < 2; ++kk) {
        v16bf a = ld_frag(Tb, 64, tm, kk * 32);
        v16bf b = ld_frag(Bp, 64, tn, kk * 32);
        acc = wmma_bf16(a, b, acc);
      }
      const int col = tn + (lane & 15);
      const int rb = tm + ((lane & 16) ? 8 : 0);
#pragma unroll
      for (int r = 0; r < 8; ++r) {
        const int row = rb + r;
        if (isU) uf[row * 128 + col] = acc[r];
        else     wb[row * 128 + col] = (bf16_t)acc[r];
      }
    }
    __syncthreads();

    // ---- phase Q: intra = tril(q @ k^T) * e^{g_r}/e^{g_c} ----
    for (int t = wid; t < 16; t += 8) {
      const int tm = (t >> 2) * 16, tn = (t & 3) * 16;
      v8f acc = (v8f){0.f,0.f,0.f,0.f,0.f,0.f,0.f,0.f};
#pragma unroll
      for (int kk = 0; kk < 4; ++kk) {
        v16bf a = ld_frag(qb, 128, tm, kk * 32);
        v16bf b = ld_frag(kb, 128, tn, kk * 32);
        acc = wmma_bf16(a, b, acc);
      }
      const int col = tn + (lane & 15);
      const int rb = tm + ((lane & 16) ? 8 : 0);
      const float dinv = gei[col];
      float grow[8];
#pragma unroll
      for (int r = 0; r < 8; ++r) grow[r] = ge[rb + r];
#pragma unroll
      for (int r = 0; r < 8; ++r) {
        const int row = rb + r;
        const float v = acc[r] * grow[r] * dinv;
        intb[row * 64 + col] = (bf16_t)((col <= row) ? v : 0.f);
      }
    }
    __syncthreads();

    // ---- phase P: v' = u - w @ state ----
    for (int t = wid; t < 32; t += 8) {
      const int tm = (t >> 3) * 16, tn = (t & 7) * 16;
      v8f acc = (v8f){0.f,0.f,0.f,0.f,0.f,0.f,0.f,0.f};
#pragma unroll
      for (int kk = 0; kk < 4; ++kk) {
        v16bf a = ld_frag(wb, 128, tm, kk * 32);
        v16bf b = ld_frag(sbT, 128, tn, kk * 32);
        acc = wmma_bf16(a, b, acc);
      }
      const int col = tn + (lane & 15);
      const int rb = tm + ((lane & 16) ? 8 : 0);
#pragma unroll
      for (int r = 0; r < 8; ++r) {
        const int row = rb + r;
        vpT[col * 64 + row] = (bf16_t)(uf[row * 128 + col] - acc[r]);
      }
    }
    __syncthreads();

    // ---- phase O: out = (q @ state) * e^g + intra @ v' ----
    for (int t = wid; t < 32; t += 8) {
      const int tm = (t >> 3) * 16, tn = (t & 7) * 16;
      v8f acc = (v8f){0.f,0.f,0.f,0.f,0.f,0.f,0.f,0.f};
#pragma unroll
      for (int kk = 0; kk < 4; ++kk) {
        v16bf a = ld_frag(qb, 128, tm, kk * 32);
        v16bf b = ld_frag(sbT, 128, tn, kk * 32);
        acc = wmma_bf16(a, b, acc);
      }
      const int rb = tm + ((lane & 16) ? 8 : 0);
#pragma unroll
      for (int r = 0; r < 8; ++r) acc[r] *= ge[rb + r];
#pragma unroll
      for (int kk = 0; kk < 2; ++kk) {
        v16bf a = ld_frag(intb, 64, tm, kk * 32);
        v16bf b = ld_frag(vpT, 64, tn, kk * 32);
        acc = wmma_bf16(a, b, acc);
      }
      const int col = tn + (lane & 15);
#pragma unroll
      for (int r = 0; r < 8; ++r) {
        const int row = rb + r;
#if HAVE_TDM
        uf[row * 128 + col] = acc[r];
#else
        core[((p0 + (long)row) * HV_N + hv) * DV_N + col] = acc[r];
#endif
      }
    }
    __syncthreads();
#if HAVE_TDM
    if (wid == 0) {
      tdm_store_2d(core + ((size_t)p0 * HV_N + hv) * DV_N, (unsigned)(size_t)uf, 2,
                   128, 64, 128u, 64u, (unsigned long long)(HV_N * DV_N));
    }
#endif

    // ---- phase S: state = state * e^{g_last} + (k*kd)^T @ v' ----
    {
      const float gel = scal[0];
      for (int t = wid; t < 64; t += 8) {
        const int tm = (t >> 3) * 16, tn = (t & 7) * 16;
        v8f acc = (v8f){0.f,0.f,0.f,0.f,0.f,0.f,0.f,0.f};
#pragma unroll
        for (int kk = 0; kk < 2; ++kk) {
          v16bf a = ld_frag(kkdT, 64, tm, kk * 32);
          v16bf b = ld_frag(vpT, 64, tn, kk * 32);
          acc = wmma_bf16(a, b, acc);
        }
        const int col = tn + (lane & 15);
        const int rb = tm + ((lane & 16) ? 8 : 0);
#pragma unroll
        for (int r = 0; r < 8; ++r) {
          const int row = rb + r;
          const int idx = row * 128 + col;
          const float ns = state[idx] * gel + acc[r];
          state[idx] = ns;
          sbT[col * 128 + row] = (bf16_t)ns;
        }
      }
    }
    __syncthreads();
  }
}

// rmsnorm(core * silu(z)) * norm_weight -> bf16
__global__ void __launch_bounds__(256)
norm_silu_kernel(const float* __restrict__ core, const bf16_t* __restrict__ qkvz,
                 const float* __restrict__ nw, bf16_t* __restrict__ outb) {
  __shared__ float red[256];
  const int tid = threadIdx.x;
  const int rloc = tid >> 5, lane = tid & 31;
  const long rowid = (long)blockIdx.x * 8 + rloc;
  const long pos = rowid >> 5;
  const int hv = (int)(rowid & 31);
  const int hk = hv >> 1, gi = hv & 1;
  const int d0 = lane * 4;
  const float4 cv = *(const float4*)(core + rowid * 128 + d0);
  union { int2 v; bf16_t h[4]; } zu;
  zu.v = *(const int2*)(qkvz + pos * QKVZ_N + hk * 768 + 512 + gi * 128 + d0);
  float y[4], ss = 0.f;
  const float cf[4] = {cv.x, cv.y, cv.z, cv.w};
#pragma unroll
  for (int j = 0; j < 4; ++j) {
    const float z = (float)zu.h[j];
    y[j] = cf[j] * (z * sigmoidf_(z));
    ss += y[j] * y[j];
  }
  red[tid] = ss;
  __syncthreads();
  if (lane == 0) {
    float s = 0.f;
    for (int t = 0; t < 32; ++t) s += red[rloc * 32 + t];
    red[rloc * 32] = rsqrtf(s * (1.f / 128.f) + 1e-6f);
  }
  __syncthreads();
  const float rn = red[rloc * 32];
  union { int2 v; bf16_t h[4]; } ou;
#pragma unroll
  for (int j = 0; j < 4; ++j) ou.h[j] = (bf16_t)(y[j] * rn * nw[d0 + j]);
  *(int2*)(outb + rowid * 128 + d0) = ou.v;
}

// ---------------- host launch ----------------
extern "C" void kernel_launch(void* const* d_in, const int* in_sizes, int n_in,
                              void* d_out, int out_size, void* d_ws, size_t ws_size,
                              hipStream_t stream) {
  (void)in_sizes; (void)n_in; (void)out_size; (void)ws_size;
  const float* x     = (const float*)d_in[0];
  const float* Wqkvz = (const float*)d_in[1];
  const float* Wba   = (const float*)d_in[2];
  const float* convw = (const float*)d_in[3];
  const float* A_log = (const float*)d_in[4];
  const float* dtb   = (const float*)d_in[5];
  const float* nw    = (const float*)d_in[6];
  const float* Wout  = (const float*)d_in[7];
  float* out = (float*)d_out;

  char* ws = (char*)d_ws;
  size_t off = 0;
  auto take = [&](size_t bytes) -> char* {
    char* p = ws + off;
    off += (bytes + 255) & ~(size_t)255;
    return p;
  };
  bf16_t* xb     = (bf16_t*)take((size_t)NTOK * D_SZ * 2);
  bf16_t* wqb    = (bf16_t*)take((size_t)D_SZ * QKVZ_N * 2);
  bf16_t* wbab   = (bf16_t*)take((size_t)D_SZ * 64 * 2);
  bf16_t* woutb  = (bf16_t*)take((size_t)VALDIM * D_SZ * 2);
  bf16_t* qkvzb  = (bf16_t*)take((size_t)NTOK * QKVZ_N * 2);
  float*  baf    = (float*)take((size_t)NTOK * 64 * 4);
  bf16_t* qkvc   = (bf16_t*)take((size_t)NTOK * CONV_N * 2);
  float*  betab  = (float*)take((size_t)NTOK * HV_N * 4);
  float*  gb     = (float*)take((size_t)NTOK * HV_N * 4);
  float*  coref  = (float*)take((size_t)NTOK * VALDIM * 4);
  bf16_t* corenb = (bf16_t*)take((size_t)NTOK * VALDIM * 2);

  f32_to_bf16_kernel<<<4096, 256, 0, stream>>>(x, xb, (long)NTOK * D_SZ);
  f32_to_bf16_kernel<<<4096, 256, 0, stream>>>(Wqkvz, wqb, (long)D_SZ * QKVZ_N);
  f32_to_bf16_kernel<<<256, 256, 0, stream>>>(Wba, wbab, (long)D_SZ * 64);
  f32_to_bf16_kernel<<<4096, 256, 0, stream>>>(Wout, woutb, (long)VALDIM * D_SZ);

  dim3 g1(QKVZ_N / 128, NTOK / 128);
  gemm_bf16_kernel<true><<<g1, 256, 0, stream>>>(xb, wqb, nullptr, qkvzb, NTOK, QKVZ_N, D_SZ);
  dim3 g2(1, NTOK / 128);
  gemm_bf16_kernel<false><<<g2, 256, 0, stream>>>(xb, wbab, baf, nullptr, NTOK, 64, D_SZ);

  const long nc8 = (long)NTOK * (CONV_N / 8);
  conv_silu_kernel<<<(unsigned)((nc8 + 255) / 256), 256, 0, stream>>>(qkvzb, convw, qkvc);

  const long nbg = (long)NTOK * HV_N;
  betag_kernel<<<(unsigned)((nbg + 255) / 256), 256, 0, stream>>>(baf, A_log, dtb, betab, gb);

  const size_t delta_lds =
      (128 * 128 + 64 * 64 + 64 * 128 + 5 * 64 + 256 + 8) * sizeof(float) +
      (128 * 128 + 8 * 64 * 128 + 3 * 64 * 64 + 2 * 4 * 16 * 32) * sizeof(bf16_t);
  delta_core_kernel<<<B_SZ * HV_N, 256, delta_lds, stream>>>(qkvc, betab, gb, coref);

  norm_silu_kernel<<<(NTOK * HV_N) / 8, 256, 0, stream>>>(coref, qkvzb, nw, corenb);

  dim3 g3(D_SZ / 128, NTOK / 128);
  gemm_bf16_kernel<true><<<g3, 256, 0, stream>>>(corenb, woutb, out, nullptr, NTOK, D_SZ, VALDIM);
}
